// GraphTransLayer_64252710748240
// MI455X (gfx1250) — matrix operand order repaired and checked by reading, not compiled, not run on hardware
//
#include <hip/hip_runtime.h>

typedef __bf16 bf16;
typedef __attribute__((ext_vector_type(16))) __bf16 v16bf;
typedef __attribute__((ext_vector_type(8)))  __bf16 v8bf;
typedef __attribute__((ext_vector_type(8)))  float  v8f;

#define DMODEL 512
#define DFF    1024
#define NH     8
#define DH     64
#define NTOK   16384
#define NB     32
#define SEQ    768
#define LSEQ   769            // SEQ + 1
#define LP     800            // LSEQ rounded up to multiple of 32
#define MROWS  (NB * LSEQ)    // 24608 (multiple of 16)
#define MR     (MROWS + 64)   // slack rows for tile overreach
#define MH     (NTOK + NB)    // 16416 (multiple of 16)
#define MHR    (MH + 64)

// ---------------- WMMA fragment helpers (CDNA5 wave32 layouts) ----------------

static __device__ inline v8f wmma_bf16(v16bf a, v16bf b, v8f c) {
  // D = A(16x32 bf16) * B(32x16 bf16) + C(f32)
  return __builtin_amdgcn_wmma_f32_16x16x32_bf16(false, a, false, b, (short)0, c,
                                                 false, false);
}

// A 16x32 bf16, row-major source with leading dim ld.
// lanes 0-15: row m=lane, elems 0..7 = K0..7, 8..15 = K16..23
// lanes 16-31: row m=lane-16, elems 0..7 = K8..15, 8..15 = K24..31
static __device__ inline v16bf load_a_g(const bf16* p, int ld) {
  const int lane = threadIdx.x & 31;
  const int m = lane & 15;
  const bf16* r = p + (size_t)m * ld + ((lane >> 4) << 3);
  v8bf q0 = *(const v8bf*)(r);
  v8bf q1 = *(const v8bf*)(r + 16);
  return __builtin_shufflevector(q0, q1, 0, 1, 2, 3, 4, 5, 6, 7,
                                 8, 9, 10, 11, 12, 13, 14, 15);
}

// B 32x16 bf16 from a transposed (N x K row-major) source: same k order as A
// (half h: elems 0..7 = K8h..8h+7, 8..15 = K16+8h..) -> two 16B vector loads per lane.
static __device__ inline v16bf load_bt(const bf16* p, int ld) {
  const int lane = threadIdx.x & 31;
  const int n = lane & 15;
  const bf16* r = p + (size_t)n * ld + ((lane >> 4) << 3);
  v8bf q0 = *(const v8bf*)(r);
  v8bf q1 = *(const v8bf*)(r + 16);
  return __builtin_shufflevector(q0, q1, 0, 1, 2, 3, 4, 5, 6, 7,
                                 8, 9, 10, 11, 12, 13, 14, 15);
}

// ---------------- small setup kernels ----------------

__global__ void init_kernel(int* row_of, int* counts) {
  const int i = blockIdx.x * blockDim.x + threadIdx.x;
  if (i < MROWS) row_of[i] = -1;
  if (i < NB) counts[i] = 0;
}

__global__ void hist_kernel(const int* __restrict__ batch, int* counts) {
  const int i = blockIdx.x * blockDim.x + threadIdx.x;
  if (i < NTOK) atomicAdd(&counts[batch[i]], 1);
}

__global__ void scan_kernel(const int* __restrict__ counts, int* starts) {
  if (threadIdx.x == 0 && blockIdx.x == 0) {
    int acc = 0;
    for (int b = 0; b < NB; ++b) { starts[b] = acc; acc += counts[b]; }
  }
}

__global__ void pos_kernel(const int* __restrict__ batch,
                           const int* __restrict__ starts, int* row_of) {
  const int i = blockIdx.x * blockDim.x + threadIdx.x;
  if (i < NTOK) {
    const int b = batch[i];
    row_of[b * LSEQ + (i - starts[b])] = i;
  }
}

__global__ void build_xd_kernel(const float* __restrict__ x,
                                const float* __restrict__ cls,
                                const int* __restrict__ row_of,
                                float* __restrict__ xf, bf16* __restrict__ xb) {
  const size_t idx = (size_t)blockIdx.x * blockDim.x + threadIdx.x;
  if (idx >= (size_t)MROWS * DMODEL) return;
  const int d = (int)(idx % DMODEL);
  const size_t r = idx / DMODEL;
  const int l = (int)(r % LSEQ);
  const int b = (int)(r / LSEQ);
  float v;
  if (l == SEQ) {
    v = cls[(size_t)b * DMODEL + d];
  } else {
    const int i = row_of[r];
    v = (i >= 0) ? x[(size_t)i * DMODEL + d] : 0.0f;
  }
  xf[idx] = v;
  xb[idx] = (bf16)v;
}

// W (K x N f32) -> Wt (N x K bf16)
__global__ void cvt_w_kernel(const float* __restrict__ W, bf16* __restrict__ Wt,
                             int Kd, int Nd) {
  const int idx = blockIdx.x * blockDim.x + threadIdx.x;
  if (idx >= Kd * Nd) return;
  const int k = idx / Nd, n = idx % Nd;
  Wt[(size_t)n * Kd + k] = (bf16)W[idx];
}

// V (token-major bf16) -> per-head transposed, zero-padded:
// Vt[((b*NH+h)*DH + d) * LP + key]
__global__ void vt_kernel(const bf16* __restrict__ V, bf16* __restrict__ Vt) {
  const size_t idx = (size_t)blockIdx.x * blockDim.x + threadIdx.x;
  if (idx >= (size_t)NB * NH * DH * LP) return;
  const int key = (int)(idx % LP);
  const int d = (int)((idx / LP) % DH);
  const int h = (int)((idx / ((size_t)LP * DH)) % NH);
  const int b = (int)(idx / ((size_t)LP * DH * NH));
  bf16 v = (bf16)0.0f;
  if (key < LSEQ) v = V[((size_t)b * LSEQ + key) * DMODEL + h * DH + d];
  Vt[idx] = v;
}

// ---------------- generic WMMA GEMM ----------------
// C[M,N] = A[M,K](bf16) * Bt[N,K](bf16)^T + bias (+resid) (ReLU?) -> f32/bf16
// block = 128 threads = 4 waves; each wave owns a 16x64 tile; block tile 16x256.
// K-loop unrolled x2 with ping-pong fragment buffers: loads land in fresh
// registers (no copy rotation) and each WMMA group overlaps the other
// buffer's in-flight loads. Kd must be a multiple of 64 (512/1024 here).
__global__ __launch_bounds__(128) void gemm_kernel(
    const bf16* __restrict__ A, int lda, const bf16* __restrict__ Bt, int ldb,
    const float* __restrict__ bias, const float* __restrict__ resid, int ldr,
    float* __restrict__ Cf, bf16* __restrict__ Cb, int ldc, int M, int N,
    int Kd, int relu) {
  const int wave = threadIdx.x >> 5;
  const int lane = threadIdx.x & 31;
  const int m0 = blockIdx.x * 16;
  const int n0 = blockIdx.y * 256 + wave * 64;
  if (m0 >= M || n0 >= N) return;
  const bf16* Ap = A + (size_t)m0 * lda;
  const bf16* Bp = Bt + (size_t)n0 * ldb;
  v8f acc[4] = {};
  // buffer A primed with k = 0
  v16bf aA = load_a_g(Ap, lda);
  v16bf bA[4];
#pragma unroll
  for (int t = 0; t < 4; ++t) bA[t] = load_bt(Bp + (size_t)(t * 16) * ldb, ldb);
  for (int k = 0; k < Kd; k += 64) {
    // issue buffer-B loads (k+32), then consume buffer A
    v16bf aB = load_a_g(Ap + k + 32, lda);
    v16bf bB[4];
#pragma unroll
    for (int t = 0; t < 4; ++t)
      bB[t] = load_bt(Bp + (size_t)(t * 16) * ldb + k + 32, ldb);
    __builtin_prefetch(Ap + k + 128, 0, 1);  // global_prefetch_b8
#pragma unroll
    for (int t = 0; t < 4; ++t) acc[t] = wmma_bf16(aA, bA[t], acc[t]);
    // issue buffer-A loads (k+64, wrap on last iter), then consume buffer B
    const int k2 = (k + 64 < Kd) ? (k + 64) : 0;
    aA = load_a_g(Ap + k2, lda);
#pragma unroll
    for (int t = 0; t < 4; ++t)
      bA[t] = load_bt(Bp + (size_t)(t * 16) * ldb + k2, ldb);
#pragma unroll
    for (int t = 0; t < 4; ++t) acc[t] = wmma_bf16(aB, bB[t], acc[t]);
  }
  const int n_in = lane & 15;
  const int mb = (lane >> 4) << 3;
#pragma unroll
  for (int t = 0; t < 4; ++t) {
    const int ncol = n0 + t * 16 + n_in;
    const float bv = bias ? bias[ncol] : 0.0f;
#pragma unroll
    for (int r = 0; r < 8; ++r) {
      const int row = m0 + mb + r;
      float v = acc[t][r] + bv;
      if (resid) v += resid[(size_t)row * ldr + ncol];
      if (relu) v = fmaxf(v, 0.0f);
      if (Cf) Cf[(size_t)row * ldc + ncol] = v;
      if (Cb) Cb[(size_t)row * ldc + ncol] = (bf16)v;
    }
  }
}

// ---------------- attention: one block per (b, h, 16-query tile) ----------------
__global__ __launch_bounds__(256) void attn_kernel(
    const bf16* __restrict__ Q, const bf16* __restrict__ Km,
    const bf16* __restrict__ Vt, const int* __restrict__ counts,
    bf16* __restrict__ O) {
  __shared__ bf16 sS[16 * LP];  // scores
  __shared__ bf16 sP[16 * LP];  // softmax probabilities
  const int bh = blockIdx.x;
  const int b = bh >> 3, h = bh & 7;
  const int qt = blockIdx.y;
  const int wave = threadIdx.x >> 5;
  const int lane = threadIdx.x & 31;
  const int q0 = qt * 16;
  const size_t rowbase = (size_t)b * LSEQ;
  const bf16* Qp = Q + (rowbase + q0) * DMODEL + h * DH;
  v16bf a0 = load_a_g(Qp, DMODEL);        // dh 0..31
  v16bf a1 = load_a_g(Qp + 32, DMODEL);   // dh 32..63
  const int cnt = counts[b];
  const int nkt = (LSEQ + 15) >> 4;       // 49 key tiles
  // 1) scores = Q K^T / sqrt(dh) + mask, tiles distributed over 8 waves
  for (int kt = wave; kt < nkt; kt += 8) {
    const bf16* Kp = Km + (rowbase + kt * 16) * DMODEL + h * DH;
    v8f acc = {};
    acc = wmma_bf16(a0, load_bt(Kp, DMODEL), acc);
    acc = wmma_bf16(a1, load_bt(Kp + 32, DMODEL), acc);
    const int n = lane & 15;
    const int mb = (lane >> 4) << 3;
    const int key = kt * 16 + n;
    const float maskv = (key < cnt || key == LSEQ - 1) ? 0.0f : -1.0e9f;
#pragma unroll
    for (int r = 0; r < 8; ++r)
      sS[(mb + r) * LP + key] = (bf16)(acc[r] * 0.125f + maskv);
  }
  __syncthreads();
  // 2) softmax: wave w owns rows w and w+8 (wave32 shuffle reductions)
#pragma unroll
  for (int rr = 0; rr < 2; ++rr) {
    const int row = wave + rr * 8;
    float mx = -3.0e38f;
    for (int c = lane; c < LSEQ; c += 32) mx = fmaxf(mx, (float)sS[row * LP + c]);
    for (int off = 16; off; off >>= 1) mx = fmaxf(mx, __shfl_xor(mx, off, 32));
    float sum = 0.0f;
    for (int c = lane; c < LSEQ; c += 32) sum += __expf((float)sS[row * LP + c] - mx);
    for (int off = 16; off; off >>= 1) sum += __shfl_xor(sum, off, 32);
    const float inv = 1.0f / sum;
    for (int c = lane; c < LP; c += 32) {
      const float pv = (c < LSEQ) ? __expf((float)sS[row * LP + c] - mx) * inv : 0.0f;
      sP[row * LP + c] = (bf16)pv;
    }
  }
  __syncthreads();
  // 3) O = P @ V : waves 0..3 each own a 16-wide slice of dh.
  // Vt is per-head transposed (d-major, key-contiguous, zero padded), so the
  // B fragment is two contiguous 16B loads per lane (same as the weight path).
  if (wave < 4) {
    const int nb = wave * 16;
    const bf16* Vtp = Vt + ((size_t)(b * NH + h) * DH + nb) * LP;
    v8f acc = {};
    for (int kk = 0; kk < LP; kk += 32) {
      v16bf a = load_a_g(sP + kk, LP);        // P fragment straight from LDS
      acc = wmma_bf16(a, load_bt(Vtp + kk, LP), acc);
    }
    const int n = lane & 15;
    const int mb = (lane >> 4) << 3;
#pragma unroll
    for (int r = 0; r < 8; ++r) {
      const int q = q0 + mb + r;
      if (q < LSEQ)
        O[(rowbase + q) * DMODEL + h * DH + nb + n] = (bf16)acc[r];
    }
  }
}

// ---------------- LayerNorm (optionally fused with un-scatter gather) ----------------

static __device__ inline float block_reduce_sum(float v) {
  __shared__ float sh[9];
  const int lane = threadIdx.x & 31;
  const int w = threadIdx.x >> 5;
  __syncthreads();
  for (int off = 16; off; off >>= 1) v += __shfl_xor(v, off, 32);
  if (lane == 0) sh[w] = v;
  __syncthreads();
  if (w == 0) {
    float t = (lane < (int)(blockDim.x >> 5)) ? sh[lane] : 0.0f;
    for (int off = 16; off; off >>= 1) t += __shfl_xor(t, off, 32);
    if (lane == 0) sh[8] = t;
  }
  __syncthreads();
  return sh[8];
}

__global__ __launch_bounds__(256) void ln_kernel(
    const float* __restrict__ in, int ldin, const int* __restrict__ batch,
    const int* __restrict__ starts, const float* __restrict__ g,
    const float* __restrict__ be, float* __restrict__ outf,
    bf16* __restrict__ outb, int ldo, int gather) {
  const int row = blockIdx.x;
  const float* xr;
  if (gather) {
    int src;
    if (row < NTOK) {
      const int b = batch[row];
      src = b * LSEQ + (row - starts[b]);
    } else {
      src = (row - NTOK) * LSEQ + (LSEQ - 1);  // CLS rows
    }
    xr = in + (size_t)src * ldin;
  } else {
    xr = in + (size_t)row * ldin;
  }
  float s = 0.0f;
  for (int c = threadIdx.x; c < DMODEL; c += blockDim.x) s += xr[c];
  const float mean = block_reduce_sum(s) * (1.0f / DMODEL);
  float s2 = 0.0f;
  for (int c = threadIdx.x; c < DMODEL; c += blockDim.x) {
    const float d = xr[c] - mean;
    s2 += d * d;
  }
  const float inv = rsqrtf(block_reduce_sum(s2) * (1.0f / DMODEL) + 1e-5f);
  for (int c = threadIdx.x; c < DMODEL; c += blockDim.x) {
    const float v = (xr[c] - mean) * inv * g[c] + be[c];
    if (outf) outf[(size_t)row * ldo + c] = v;
    if (outb) outb[(size_t)row * ldo + c] = (bf16)v;
  }
}

// ---------------- host-side orchestration ----------------

extern "C" void kernel_launch(void* const* d_in, const int* in_sizes, int n_in,
                              void* d_out, int out_size, void* d_ws,
                              size_t ws_size, hipStream_t stream) {
  (void)in_sizes; (void)n_in; (void)out_size; (void)ws_size;
  const float* x     = (const float*)d_in[0];
  const int*   batch = (const int*)d_in[1];
  const float* cls   = (const float*)d_in[2];
  const float* Wq = (const float*)d_in[3];  const float* bq = (const float*)d_in[4];
  const float* Wk = (const float*)d_in[5];  const float* bk = (const float*)d_in[6];
  const float* Wv = (const float*)d_in[7];  const float* bv = (const float*)d_in[8];
  const float* Wo = (const float*)d_in[9];  const float* bo = (const float*)d_in[10];
  const float* W1 = (const float*)d_in[11]; const float* b1 = (const float*)d_in[12];
  const float* W2 = (const float*)d_in[13]; const float* b2 = (const float*)d_in[14];
  const float* g0 = (const float*)d_in[15]; const float* be0 = (const float*)d_in[16];
  const float* g1 = (const float*)d_in[17]; const float* be1 = (const float*)d_in[18];

  char* wsp = (char*)d_ws;
  auto alloc = [&](size_t bytes) -> void* {
    void* p = (void*)wsp;
    wsp += (bytes + 255) & ~(size_t)255;
    return p;
  };
  int*   counts = (int*)alloc(NB * sizeof(int));
  int*   starts = (int*)alloc(NB * sizeof(int));
  int*   row_of = (int*)alloc((size_t)MROWS * sizeof(int));
  float* xdf = (float*)alloc((size_t)MR * DMODEL * 4);
  bf16*  xdb = (bf16*)alloc((size_t)MR * DMODEL * 2);
  bf16*  Wqt = (bf16*)alloc((size_t)DMODEL * DMODEL * 2);
  bf16*  Wkt = (bf16*)alloc((size_t)DMODEL * DMODEL * 2);
  bf16*  Wvt = (bf16*)alloc((size_t)DMODEL * DMODEL * 2);
  bf16*  Wot = (bf16*)alloc((size_t)DMODEL * DMODEL * 2);
  bf16*  W1t = (bf16*)alloc((size_t)DFF * DMODEL * 2);     // 1024 x 512
  bf16*  W2t = (bf16*)alloc((size_t)DMODEL * DFF * 2);     // 512 x 1024
  bf16*  Qb = (bf16*)alloc((size_t)MR * DMODEL * 2);
  bf16*  Kb = (bf16*)alloc((size_t)MR * DMODEL * 2);
  bf16*  Vb = (bf16*)alloc((size_t)MR * DMODEL * 2);
  bf16*  Vtb = (bf16*)alloc((size_t)NB * NH * DH * LP * 2);
  bf16*  attnb = (bf16*)alloc((size_t)MR * DMODEL * 2);
  float* aof = (float*)alloc((size_t)MR * DMODEL * 4);
  float* hlf = (float*)alloc((size_t)MHR * DMODEL * 4);
  bf16*  hlb = (bf16*)alloc((size_t)MHR * DMODEL * 2);
  bf16*  f1b = (bf16*)alloc((size_t)MHR * DFF * 2);
  float* tmpf = (float*)alloc((size_t)MHR * DMODEL * 4);

  // segment bookkeeping + dense padded layout
  init_kernel<<<(MROWS + 255) / 256, 256, 0, stream>>>(row_of, counts);
  hist_kernel<<<(NTOK + 255) / 256, 256, 0, stream>>>(batch, counts);
  scan_kernel<<<1, 32, 0, stream>>>(counts, starts);
  pos_kernel<<<(NTOK + 255) / 256, 256, 0, stream>>>(batch, starts, row_of);
  build_xd_kernel<<<(int)(((size_t)MROWS * DMODEL + 255) / 256), 256, 0, stream>>>(
      x, cls, row_of, xdf, xdb);

  // weights -> bf16, transposed (N x K)
  cvt_w_kernel<<<(DMODEL * DMODEL + 255) / 256, 256, 0, stream>>>(Wq, Wqt, DMODEL, DMODEL);
  cvt_w_kernel<<<(DMODEL * DMODEL + 255) / 256, 256, 0, stream>>>(Wk, Wkt, DMODEL, DMODEL);
  cvt_w_kernel<<<(DMODEL * DMODEL + 255) / 256, 256, 0, stream>>>(Wv, Wvt, DMODEL, DMODEL);
  cvt_w_kernel<<<(DMODEL * DMODEL + 255) / 256, 256, 0, stream>>>(Wo, Wot, DMODEL, DMODEL);
  cvt_w_kernel<<<(DMODEL * DFF + 255) / 256, 256, 0, stream>>>(W1, W1t, DMODEL, DFF);
  cvt_w_kernel<<<(DFF * DMODEL + 255) / 256, 256, 0, stream>>>(W2, W2t, DFF, DMODEL);

  // Q, K, V projections (bf16 out)
  dim3 gq(MROWS / 16, DMODEL / 256);
  gemm_kernel<<<gq, 128, 0, stream>>>(xdb, DMODEL, Wqt, DMODEL, bq, nullptr, 0,
                                      nullptr, Qb, DMODEL, MROWS, DMODEL, DMODEL, 0);
  gemm_kernel<<<gq, 128, 0, stream>>>(xdb, DMODEL, Wkt, DMODEL, bk, nullptr, 0,
                                      nullptr, Kb, DMODEL, MROWS, DMODEL, DMODEL, 0);
  gemm_kernel<<<gq, 128, 0, stream>>>(xdb, DMODEL, Wvt, DMODEL, bv, nullptr, 0,
                                      nullptr, Vb, DMODEL, MROWS, DMODEL, DMODEL, 0);

  // per-head transposed zero-padded V for the P@V WMMA path
  vt_kernel<<<(int)(((size_t)NB * NH * DH * LP + 255) / 256), 256, 0, stream>>>(Vb, Vtb);

  // attention
  attn_kernel<<<dim3(NB * NH, (LSEQ + 15) / 16), 256, 0, stream>>>(Qb, Kb, Vtb,
                                                                   counts, attnb);

  // output projection + residual (xd)
  gemm_kernel<<<gq, 128, 0, stream>>>(attnb, DMODEL, Wot, DMODEL, bo, xdf, DMODEL,
                                      aof, nullptr, DMODEL, MROWS, DMODEL, DMODEL, 0);

  // gather back to token order (+CLS rows) fused with LayerNorm 0
  ln_kernel<<<MH, 256, 0, stream>>>(aof, DMODEL, batch, starts, g0, be0, hlf, hlb,
                                    DMODEL, 1);

  // FFN: relu(h W1 + b1) W2 + b2 + h
  gemm_kernel<<<dim3(MH / 16, DFF / 256), 128, 0, stream>>>(
      hlb, DMODEL, W1t, DMODEL, b1, nullptr, 0, nullptr, f1b, DFF, MH, DFF, DMODEL, 1);
  gemm_kernel<<<dim3(MH / 16, DMODEL / 256), 128, 0, stream>>>(
      f1b, DFF, W2t, DFF, b2, hlf, DMODEL, tmpf, nullptr, DMODEL, MH, DMODEL, DFF, 0);

  // final LayerNorm -> output
  ln_kernel<<<MH, 256, 0, stream>>>(tmpf, DMODEL, nullptr, nullptr, g1, be1,
                                    (float*)d_out, nullptr, DMODEL, 0);
}
